// CausalSelfAttention_33320356282737
// MI455X (gfx1250) — compile-verified, matrix operand order checked by reading, not run on hardware
//
#include <hip/hip_runtime.h>
#include <hip/hip_bf16.h>
#include <stdint.h>

typedef __bf16 bf16;
typedef __bf16 v8bf  __attribute__((ext_vector_type(8)));
typedef __bf16 v16bf __attribute__((ext_vector_type(16)));
typedef float  v8f   __attribute__((ext_vector_type(8)));

#define N_EMBD 1024
#define N_HEAD 16
#define HS     64
#define T_SEQ  2048
#define BATCH  2
#define M_TOT  (BATCH * T_SEQ)   // 4096

// Async global->LDS copy, 16B per lane. ASYNCcnt-tracked (CDNA5).
// NOTE: INST_OFFSET is added to BOTH the global and the LDS address, so the
// chunks must be contiguous on both sides (they are, within one LDS row).
#define ASYNC_LD_B128(ldsaddr, voff, sbase, IMM)                              \
    asm volatile("global_load_async_to_lds_b128 %0, %1, %2 offset:" IMM      \
                 :: "v"(ldsaddr), "v"(voff), "s"(sbase) : "memory")

__device__ __forceinline__ unsigned lds_u32(const void* p) {
    // Low 32 bits of the flat address of a __shared__ object = LDS byte addr.
    return (unsigned)(uintptr_t)p;
}

// ---------------------------------------------------------------------------
// DPP16 16-lane butterfly reductions (pure VALU, no LDS, no waits).
// Lanes 0-15 and 16-31 reduce independently == exactly the C/D row groups.
// ---------------------------------------------------------------------------
template <int CTRL>
__device__ __forceinline__ float dppf(float x) {
    int xi = __builtin_bit_cast(int, x);
    int r  = __builtin_amdgcn_update_dpp(xi, xi, CTRL, 0xF, 0xF, true);
    return __builtin_bit_cast(float, r);
}
__device__ __forceinline__ float red16_max(float x) {
    x = fmaxf(x, dppf<0xB1>(x));    // quad_perm(1,0,3,2)
    x = fmaxf(x, dppf<0x4E>(x));    // quad_perm(2,3,0,1)
    x = fmaxf(x, dppf<0x141>(x));   // row_half_mirror
    x = fmaxf(x, dppf<0x140>(x));   // row_mirror
    return x;
}
__device__ __forceinline__ float red16_sum(float x) {
    x += dppf<0xB1>(x);
    x += dppf<0x4E>(x);
    x += dppf<0x141>(x);
    x += dppf<0x140>(x);
    return x;
}

// ---------------------------------------------------------------------------
// Fragment gather from a k-major tile.
// CDNA5 16-bit A/B layout: lanes 0-15 hold row (A) / col (B) = lane&15,
// halves map to K = {c..c+7} U {c+16..c+23}, c = 8*(lane>=16)
// -> exactly two contiguous 16-byte loads per fragment.
// ---------------------------------------------------------------------------
__device__ __forceinline__ v16bf frag_ld(const bf16* base, int ld, int lane) {
    const int g = (lane >> 4) & 1;
    const int r = lane & 15;
    const bf16* p = base + r * ld + g * 8;
    v8bf lo = *(const v8bf*)(p);
    v8bf hi = *(const v8bf*)(p + 16);
    return __builtin_shufflevector(lo, hi, 0,1,2,3,4,5,6,7,8,9,10,11,12,13,14,15);
}

__device__ __forceinline__ v16bf frag_ld_global(const bf16* rowptr, int lane, int koff) {
    const int g = (lane >> 4) & 1;
    const bf16* p = rowptr + koff + g * 8;
    v8bf lo = *(const v8bf*)(p);
    v8bf hi = *(const v8bf*)(p + 16);
    return __builtin_shufflevector(lo, hi, 0,1,2,3,4,5,6,7,8,9,10,11,12,13,14,15);
}

// ---------------------------------------------------------------------------
// fp32 -> bf16 conversion
// ---------------------------------------------------------------------------
__global__ __launch_bounds__(256) void cvt_f32_bf16(const float* __restrict__ in,
                                                    bf16* __restrict__ out, int n) {
    int i = (blockIdx.x * blockDim.x + threadIdx.x) * 4;
    if (i + 3 < n) {
        float4 v = *(const float4*)(in + i);
        out[i + 0] = (bf16)v.x;
        out[i + 1] = (bf16)v.y;
        out[i + 2] = (bf16)v.z;
        out[i + 3] = (bf16)v.w;
    }
}

// ---------------------------------------------------------------------------
// GEMM: Out[M,N] = A[M,K] @ W[N,K]^T + bias   (A, W bf16; acc fp32)
// 256 thr (8 waves). Tile 128x128, BK=64, double-buffered async-to-LDS.
// Wave -> 64x32 (2x4 wave grid), 4x2 16x16 accums -> 16 WMMAs per K-step.
// MODE 0: store bf16 packed to (B, NH, T, HS)   (QKV path)
// MODE 1: store fp32 to row-major [M, N]        (output projection)
// ---------------------------------------------------------------------------
template <int MODE>
__global__ __launch_bounds__(256)
void gemm_bf16(const bf16* __restrict__ A, const bf16* __restrict__ W,
               const float* __restrict__ bias, void* __restrict__ Out) {
    constexpr int K   = N_EMBD;
    constexpr int BK  = 64;
    constexpr int LDT = BK + 8;            // 72 halves = 144B rows (16B aligned)
    __shared__ bf16 As[2][128 * LDT];
    __shared__ bf16 Ws[2][128 * LDT];

    const int tid  = threadIdx.x;
    const int lane = tid & 31;
    const int w    = tid >> 5;
    const int wm   = w >> 2;               // 0..1 : M offset * 64
    const int wn   = w & 3;                // 0..3 : N offset * 32
    const int rowBase = blockIdx.y * 128;
    const int colBase = blockIdx.x * 128;

    v8f acc[4][2];
#pragma unroll
    for (int mi = 0; mi < 4; mi++)
#pragma unroll
        for (int ni = 0; ni < 2; ni++) acc[mi][ni] = (v8f)0.0f;

    // loader: each thread owns one 64B half-row per array per tile
    const int ldr = tid >> 1;              // 0..127
    const int ldc = (tid & 1) * 32;        // 0 or 32 halves
    const uint64_t abase = (uint64_t)(uintptr_t)A;
    const uint64_t wbase = (uint64_t)(uintptr_t)W;
    const unsigned aoff0 = (unsigned)(((size_t)(rowBase + ldr) * K + ldc) * sizeof(bf16));
    const unsigned woff0 = (unsigned)(((size_t)(colBase + ldr) * K + ldc) * sizeof(bf16));
    unsigned aDst[2], wDst[2];
#pragma unroll
    for (int b = 0; b < 2; b++) {
        aDst[b] = lds_u32(&As[b][ldr * LDT + ldc]);
        wDst[b] = lds_u32(&Ws[b][ldr * LDT + ldc]);
    }

    auto issue_tile = [&](int buf, int k0) {
        const unsigned av = aoff0 + (unsigned)(k0 * sizeof(bf16));
        ASYNC_LD_B128(aDst[buf], av, abase, "0");
        ASYNC_LD_B128(aDst[buf], av, abase, "16");
        ASYNC_LD_B128(aDst[buf], av, abase, "32");
        ASYNC_LD_B128(aDst[buf], av, abase, "48");
        const unsigned wv = woff0 + (unsigned)(k0 * sizeof(bf16));
        ASYNC_LD_B128(wDst[buf], wv, wbase, "0");
        ASYNC_LD_B128(wDst[buf], wv, wbase, "16");
        ASYNC_LD_B128(wDst[buf], wv, wbase, "32");
        ASYNC_LD_B128(wDst[buf], wv, wbase, "48");
    };

    issue_tile(0, 0);
    constexpr int NST = K / BK;            // 16 steps
    for (int i = 0; i < NST; i++) {
        const int buf = i & 1;
        if (i + 1 < NST) {
            issue_tile(buf ^ 1, (i + 1) * BK);
            asm volatile("s_wait_asynccnt 0x8" ::: "memory"); // current tile done
        } else {
            asm volatile("s_wait_asynccnt 0x0" ::: "memory");
        }
        __syncthreads();
#pragma unroll
        for (int ks = 0; ks < 2; ks++) {
            v16bf bfr[2];
#pragma unroll
            for (int ni = 0; ni < 2; ni++)
                bfr[ni] = frag_ld(&Ws[buf][(wn * 32 + ni * 16) * LDT + ks * 32], LDT, lane);
#pragma unroll
            for (int mi = 0; mi < 4; mi++) {
                v16bf af = frag_ld(&As[buf][(wm * 64 + mi * 16) * LDT + ks * 32], LDT, lane);
#pragma unroll
                for (int ni = 0; ni < 2; ni++)
                    acc[mi][ni] = __builtin_amdgcn_wmma_f32_16x16x32_bf16(
                        false, af, false, bfr[ni], (short)0, acc[mi][ni], false, false);
            }
        }
        __syncthreads();
    }

    // Epilogue. C/D layout: lane group g = lane>>4, n = lane&15; VGPR r -> row r+8g.
    const int g = lane >> 4, n = lane & 15;
#pragma unroll
    for (int mi = 0; mi < 4; mi++) {
#pragma unroll
        for (int ni = 0; ni < 2; ni++) {
            const int col = colBase + wn * 32 + ni * 16 + n;
            const float bv = bias[col];
#pragma unroll
            for (int r = 0; r < 8; r++) {
                const int row = rowBase + wm * 64 + mi * 16 + r + 8 * g;
                const float val = acc[mi][ni][r] + bv;
                if (MODE == 0) {
                    const int b = row >> 11, t = row & (T_SEQ - 1);
                    const int h = col >> 6, d = col & (HS - 1);
                    ((bf16*)Out)[(((size_t)(b * N_HEAD + h) * T_SEQ + t) * HS) + d] = (bf16)val;
                } else {
                    ((float*)Out)[(size_t)row * N_EMBD + col] = val;
                }
            }
        }
    }
}

// ---------------------------------------------------------------------------
// In-place interleaved RoPE on q and k, layout (B, NH, T, HS), bf16.
// ---------------------------------------------------------------------------
__global__ __launch_bounds__(256) void rope_kernel(bf16* __restrict__ q, bf16* __restrict__ k) {
    const int i = blockIdx.x * blockDim.x + threadIdx.x;   // 0 .. B*NH*T*32-1
    const int d2 = i & 31;
    const int t  = (i >> 5) & (T_SEQ - 1);
    const int bh = i >> 16;
    const size_t base = ((size_t)bh * T_SEQ + t) * HS + 2 * d2;
    const float inv_freq = __powf(10000.0f, -(float)(2 * d2) / (float)HS);
    float s, c;
    __sincosf((float)t * inv_freq, &s, &c);
    float x0 = (float)q[base], x1 = (float)q[base + 1];
    q[base]     = (bf16)(x0 * c - x1 * s);
    q[base + 1] = (bf16)(x0 * s + x1 * c);
    x0 = (float)k[base]; x1 = (float)k[base + 1];
    k[base]     = (bf16)(x0 * c - x1 * s);
    k[base + 1] = (bf16)(x0 * s + x1 * c);
}

// ---------------------------------------------------------------------------
// Flash attention, causal. Grid (T/128, NH, B), block 256 (8 waves).
// Wave owns 16 q rows. K tiles: async global->LDS, ping-pong. V tiles:
// prefetched to VGPRs one iteration ahead, pair-packed and scattered
// (transposed, b32 stores) into the ping-pong Vt buffer. Online softmax with
// DPP16 butterfly reductions; QK^T and PV on WMMA bf16. Mask math only on
// diagonal tiles; fully-masked wave x tile combos skip compute entirely.
// ---------------------------------------------------------------------------
__global__ __launch_bounds__(256)
void attn_kernel(const bf16* __restrict__ Q, const bf16* __restrict__ Kb,
                 const bf16* __restrict__ Vb, bf16* __restrict__ Y) {
    constexpr int LDK = 72;                 // 144B rows, 16B aligned
    __shared__ bf16 Ks[2][64 * LDK];        // [key][d]
    __shared__ bf16 Vt[2][64 * LDK];        // [d][key]  (transposed)
    __shared__ bf16 Ps[8 * 16 * LDK];       // per-wave P staging [qrow][key]

    const int tid = threadIdx.x, lane = tid & 31, w = tid >> 5;
    const int g = lane >> 4, ln = lane & 15;
    const int bh = blockIdx.z * N_HEAD + blockIdx.y;
    const int qBase = blockIdx.x * 128;
    const int qRow0 = qBase + w * 16;

    const bf16* qp = Q  + (size_t)bh * T_SEQ * HS;
    const bf16* kp = Kb + (size_t)bh * T_SEQ * HS;
    const bf16* vp = Vb + (size_t)bh * T_SEQ * HS;

    // Q fragments (persist for whole kv loop)
    v16bf qf[2];
    {
        const bf16* p = qp + (size_t)(qRow0 + ln) * HS;
#pragma unroll
        for (int ks = 0; ks < 2; ks++) qf[ks] = frag_ld_global(p, lane, ks * 32);
    }

    v8f o[4];
#pragma unroll
    for (int nt = 0; nt < 4; nt++) o[nt] = (v8f)0.0f;
    float mi[8], li[8];
#pragma unroll
    for (int r = 0; r < 8; r++) { mi[r] = -1e30f; li[r] = 0.0f; }

    // K loader: thread -> (row lr, 32B chunk lc) of the 64x64 tile
    const int lr = tid >> 2, lc = (tid & 3) * 16;
    const uint64_t kbase64 = (uint64_t)(uintptr_t)kp;
    unsigned kDst[2];
    kDst[0] = lds_u32(&Ks[0][lr * LDK + lc]);
    kDst[1] = lds_u32(&Ks[1][lr * LDK + lc]);
    auto issue_k = [&](int buf, int kBase) {
        const unsigned kv = (unsigned)(((size_t)(kBase + lr) * HS + lc) * sizeof(bf16));
        ASYNC_LD_B128(kDst[buf], kv, kbase64, "0");
        ASYNC_LD_B128(kDst[buf], kv, kbase64, "16");
    };

    // V loader: thread -> 2 adjacent keys x 8 d-values -> packed b32 scatter
    const int vkey = (tid >> 3) * 2;        // 0,2,..,62
    const int vd   = (tid & 7) * 8;         // 0,8,..,56

    const float scale = 0.125f;             // 1/sqrt(64)
    const int nkv = (qBase >> 6) + 2;       // causal: key tiles 0 .. qBase/64+1

    // prologue: start tile 0
    issue_k(0, 0);
    v8bf vr0 = *(const v8bf*)(vp + (size_t)vkey * HS + vd);
    v8bf vr1 = *(const v8bf*)(vp + (size_t)(vkey + 1) * HS + vd);

    for (int kt = 0; kt < nkv; kt++) {
        const int buf = kt & 1;
        const int kBase = kt * 64;

        // transpose-scatter the prefetched V tile into Vt[buf] (packed pairs)
#pragma unroll
        for (int j = 0; j < 8; j++) {
            union { bf16 h[2]; unsigned u; } pk;
            pk.h[0] = vr0[j];               // key  vkey
            pk.h[1] = vr1[j];               // key  vkey+1
            *(unsigned*)&Vt[buf][(vd + j) * LDK + vkey] = pk.u;
        }
        asm volatile("s_wait_asynccnt 0x0" ::: "memory");  // K tile landed
        __syncthreads();

        if (kt + 1 < nkv) {                 // prefetch next tile
            issue_k(buf ^ 1, kBase + 64);
            const bf16* pv = vp + (size_t)(kBase + 64 + vkey) * HS + vd;
            vr0 = *(const v8bf*)(pv);
            vr1 = *(const v8bf*)(pv + HS);
        }

        // skip wave x tile combos that are entirely above the diagonal
        const int doWork = __builtin_amdgcn_readfirstlane(
            (kBase <= qRow0 + 15) ? 1 : 0);
        if (doWork) {
            // S = scale * q @ k^T
            v8f s[4];
#pragma unroll
            for (int nt = 0; nt < 4; nt++) s[nt] = (v8f)0.0f;
#pragma unroll
            for (int ks = 0; ks < 2; ks++) {
#pragma unroll
                for (int nt = 0; nt < 4; nt++) {
                    v16bf kf = frag_ld(&Ks[buf][(nt * 16) * LDK + ks * 32], LDK, lane);
                    s[nt] = __builtin_amdgcn_wmma_f32_16x16x32_bf16(
                        false, qf[ks], false, kf, (short)0, s[nt], false, false);
                }
            }

            // scale (+ causal mask on diagonal tiles only) + row max
            const int needMask = __builtin_amdgcn_readfirstlane(
                (kBase + 63 > qRow0) ? 1 : 0);
            float tmax[8];
#pragma unroll
            for (int r = 0; r < 8; r++) tmax[r] = -1e30f;
            if (needMask) {
#pragma unroll
                for (int nt = 0; nt < 4; nt++) {
                    const int kc = kBase + nt * 16 + ln;
#pragma unroll
                    for (int r = 0; r < 8; r++) {
                        const int qr = qRow0 + r + 8 * g;
                        float v = s[nt][r] * scale;
                        if (kc > qr) v = -1e30f;
                        s[nt][r] = v;
                        tmax[r] = fmaxf(tmax[r], v);
                    }
                }
            } else {
#pragma unroll
                for (int nt = 0; nt < 4; nt++)
#pragma unroll
                    for (int r = 0; r < 8; r++) {
                        const float v = s[nt][r] * scale;
                        s[nt][r] = v;
                        tmax[r] = fmaxf(tmax[r], v);
                    }
            }
#pragma unroll
            for (int r = 0; r < 8; r++) tmax[r] = red16_max(tmax[r]);

            // online softmax update
#pragma unroll
            for (int r = 0; r < 8; r++) {
                const float mn = fmaxf(mi[r], tmax[r]);
                const float alpha = __expf(mi[r] - mn);
                mi[r] = mn;
                li[r] *= alpha;
#pragma unroll
                for (int nt = 0; nt < 4; nt++) o[nt][r] *= alpha;
            }

            // P = exp(S - m) -> LDS staging (wave-private) + row sums
            float rs[8];
#pragma unroll
            for (int r = 0; r < 8; r++) rs[r] = 0.0f;
#pragma unroll
            for (int nt = 0; nt < 4; nt++)
#pragma unroll
                for (int r = 0; r < 8; r++) {
                    const float p = __expf(s[nt][r] - mi[r]);
                    rs[r] += p;
                    Ps[(w * 16 + r + 8 * g) * LDK + nt * 16 + ln] = (bf16)p;
                }
#pragma unroll
            for (int r = 0; r < 8; r++) li[r] += red16_sum(rs[r]);

            asm volatile("s_wait_dscnt 0x0" ::: "memory");   // wave-private staging

            // O += P @ V
#pragma unroll
            for (int ks = 0; ks < 2; ks++) {
                v16bf pf = frag_ld(&Ps[(w * 16) * LDK + ks * 32], LDK, lane);
#pragma unroll
                for (int nt = 0; nt < 4; nt++) {
                    v16bf vf = frag_ld(&Vt[buf][(nt * 16) * LDK + ks * 32], LDK, lane);
                    o[nt] = __builtin_amdgcn_wmma_f32_16x16x32_bf16(
                        false, pf, false, vf, (short)0, o[nt], false, false);
                }
            }
        }
        __syncthreads();   // protect ping-pong buffers
    }

    // normalize + write Y as (B, T, C) bf16
    const int b = blockIdx.z, h = blockIdx.y;
#pragma unroll
    for (int r = 0; r < 8; r++) {
        const float inv = 1.0f / li[r];
        const int t = qRow0 + r + 8 * g;
#pragma unroll
        for (int nt = 0; nt < 4; nt++) {
            const int col = h * HS + nt * 16 + ln;
            Y[(size_t)(b * T_SEQ + t) * N_EMBD + col] = (bf16)(o[nt][r] * inv);
        }
    }
}

// ---------------------------------------------------------------------------
extern "C" void kernel_launch(void* const* d_in, const int* in_sizes, int n_in,
                              void* d_out, int out_size, void* d_ws, size_t ws_size,
                              hipStream_t stream) {
    const float* x  = (const float*)d_in[0];
    const float* Wq = (const float*)d_in[1];
    const float* bq = (const float*)d_in[2];
    const float* Wk = (const float*)d_in[3];
    const float* bk = (const float*)d_in[4];
    const float* Wv = (const float*)d_in[5];
    const float* bv = (const float*)d_in[6];
    const float* Wp = (const float*)d_in[7];
    const float* bp = (const float*)d_in[8];

    char* ws = (char*)d_ws;
    const size_t actSz = (size_t)M_TOT * N_EMBD * sizeof(bf16);   // 8 MB
    const size_t wSz   = (size_t)N_EMBD * N_EMBD * sizeof(bf16);  // 2 MB
    bf16* xb  = (bf16*)ws;             ws += actSz;
    bf16* Wqb = (bf16*)ws;             ws += wSz;
    bf16* Wkb = (bf16*)ws;             ws += wSz;
    bf16* Wvb = (bf16*)ws;             ws += wSz;
    bf16* Wpb = (bf16*)ws;             ws += wSz;
    bf16* qb  = (bf16*)ws;             ws += actSz;
    bf16* kb_ = (bf16*)ws;             ws += actSz;
    bf16* vb_ = (bf16*)ws;             ws += actSz;
    bf16* yb  = (bf16*)ws;             ws += actSz;

    const int nAct = M_TOT * N_EMBD;        // 4,194,304
    const int nW   = N_EMBD * N_EMBD;       // 1,048,576
    cvt_f32_bf16<<<nAct / 4 / 256, 256, 0, stream>>>(x,  xb,  nAct);
    cvt_f32_bf16<<<nW   / 4 / 256, 256, 0, stream>>>(Wq, Wqb, nW);
    cvt_f32_bf16<<<nW   / 4 / 256, 256, 0, stream>>>(Wk, Wkb, nW);
    cvt_f32_bf16<<<nW   / 4 / 256, 256, 0, stream>>>(Wv, Wvb, nW);
    cvt_f32_bf16<<<nW   / 4 / 256, 256, 0, stream>>>(Wp, Wpb, nW);

    dim3 gGemm(N_EMBD / 128, M_TOT / 128);  // (8, 32)
    gemm_bf16<0><<<gGemm, 256, 0, stream>>>(xb, Wqb, bq, qb);
    gemm_bf16<0><<<gGemm, 256, 0, stream>>>(xb, Wkb, bk, kb_);
    gemm_bf16<0><<<gGemm, 256, 0, stream>>>(xb, Wvb, bv, vb_);

    const int nPairs = BATCH * N_HEAD * T_SEQ * (HS / 2);  // 2,097,152
    rope_kernel<<<nPairs / 256, 256, 0, stream>>>(qb, kb_);

    dim3 gAttn(T_SEQ / 128, N_HEAD, BATCH); // (16, 16, 2)
    attn_kernel<<<gAttn, 256, 0, stream>>>(qb, kb_, vb_, yb);

    gemm_bf16<1><<<gGemm, 256, 0, stream>>>(yb, Wpb, bp, d_out);
}
